// EdgePredictionTransferModel_3040836846243
// MI455X (gfx1250) — matrix operand order repaired and checked by reading, not compiled, not run on hardware
//
#include <hip/hip_runtime.h>
#include <hip/hip_bf16.h>
#include <math.h>

#define D 300
#define D2 600
#define NLAYERS 4
#define BN_EPS 1e-5f

typedef __attribute__((ext_vector_type(16))) _Float16 v16h;
typedef __attribute__((ext_vector_type(4)))  _Float16 v4h;
typedef __attribute__((ext_vector_type(8)))  float    v8f;

__device__ __forceinline__ void atomAddGlobal(float* p, float v) {
  __hip_atomic_fetch_add(p, v, __ATOMIC_RELAXED, __HIP_MEMORY_SCOPE_AGENT);
}
__device__ __forceinline__ void atomAddLds(float* p, float v) {
  __hip_atomic_fetch_add(p, v, __ATOMIC_RELAXED, __HIP_MEMORY_SCOPE_WORKGROUP);
}

// ---------------------------------------------------------------------------
// Tiny MLP for node/edge encoders on constant-ones input.
// ---------------------------------------------------------------------------
__global__ void encoder_kernel(const float* __restrict__ W0, const float* __restrict__ b0,
                               const float* __restrict__ W1, const float* __restrict__ b1,
                               const float* __restrict__ W2, const float* __restrict__ b2,
                               float* __restrict__ out) {
  __shared__ float t0[D];
  __shared__ float t1[D];
  int t = threadIdx.x;
  if (t < D) t0[t] = fmaxf(W0[t] + b0[t], 0.f);
  __syncthreads();
  if (t < D) {
    float s = b1[t];
    for (int k = 0; k < D; ++k) s += t0[k] * W1[k * D + t];
    t1[t] = fmaxf(s, 0.f);
  }
  __syncthreads();
  if (t < D) {
    float s = b2[t];
    for (int k = 0; k < D; ++k) s += t1[k] * W2[k * D + t];
    out[t] = s;
  }
}

__global__ void broadcast_kernel(const float* __restrict__ h0, float* __restrict__ H, int total) {
  int t = blockIdx.x * blockDim.x + threadIdx.x;
  if (t < total) H[t] = h0[t % D];
}

__global__ void copy_kernel(const float* __restrict__ A, float* __restrict__ B, int total) {
  int t = blockIdx.x * blockDim.x + threadIdx.x;
  if (t < total) B[t] = A[t];
}

__global__ void zero_kernel(float* __restrict__ A, int total) {
  int t = blockIdx.x * blockDim.x + threadIdx.x;
  if (t < total) A[t] = 0.f;
}

// ---------------------------------------------------------------------------
// GINE message passing: X[dst] += relu(H[src] + e0)   (X pre-loaded with H)
// ---------------------------------------------------------------------------
__global__ void message_kernel(const float* __restrict__ H, const float* __restrict__ e0,
                               const int* __restrict__ src, const int* __restrict__ dst,
                               float* __restrict__ X, int E) {
  int t = blockIdx.x * blockDim.x + threadIdx.x;
  const int QPR = D / 4;
  int total = E * QPR;
  if (t >= total) return;
  int e = t / QPR;
  int q = t - e * QPR;
  int s = src[e], d = dst[e];
  float4 hv = ((const float4*)(H + (size_t)s * D))[q];
  float4 ev = ((const float4*)e0)[q];
  float* out = X + (size_t)d * D + q * 4;
  float m;
  m = hv.x + ev.x; if (m > 0.f) atomAddGlobal(out + 0, m);
  m = hv.y + ev.y; if (m > 0.f) atomAddGlobal(out + 1, m);
  m = hv.z + ev.z; if (m > 0.f) atomAddGlobal(out + 2, m);
  m = hv.w + ev.w; if (m > 0.f) atomAddGlobal(out + 3, m);
}

// ---------------------------------------------------------------------------
// Pre-pack W[K,Nout] (f32 row-major) into f16 WMMA B-fragment layout:
// Wpk[((nt*KT + kt)*32 + lane)*16 + j] = W[kt*32 + 8*(lane>>4) + j + (j>=8 ? 8 : 0)][nt*16 + (lane&15)]
// so the GEMM loads each B fragment as 32 contiguous bytes (2x global_load_b128).
// ---------------------------------------------------------------------------
__global__ void pack_w_kernel(const float* __restrict__ W, _Float16* __restrict__ Wpk,
                              int K, int Nout, int KT, int NTp) {
  int t = blockIdx.x * blockDim.x + threadIdx.x;
  int total = NTp * KT * 32;
  if (t >= total) return;
  int lane = t & 31;
  int kt = (t >> 5) % KT;
  int nt = (t >> 5) / KT;
  int ncol = nt * 16 + (lane & 15);
  int kbase = kt * 32 + ((lane < 16) ? 0 : 8);
  v16h frag;
  #pragma unroll
  for (int j = 0; j < 16; ++j) {
    int k = kbase + j + ((j >= 8) ? 8 : 0);
    float v = (k < K && ncol < Nout) ? W[(size_t)k * Nout + ncol] : 0.f;
    frag[j] = (_Float16)v;
  }
  *(v16h*)(Wpk + (size_t)t * 16) = frag;
}

// ---------------------------------------------------------------------------
// WMMA GEMM: C[M,Nout] = act(A[M,K] @ W + bias), W pre-packed as fragments.
// GATHER==1: A row r = concat(Hn[idx0[r],:300], Hn[idx1[r],:300]) (K=600).
// 256 threads = 8 waves; block tile 64x64; wave (rg,ng) owns rows rg*16..+15,
// cols ng*32..+31 (2 n-tiles, 2 accumulators). A tile staged in LDS in
// fragment-swizzled order: lane reads its 16 halves contiguously (b128 x2).
// ---------------------------------------------------------------------------
template <int GATHER, bool RELU>
__global__ __launch_bounds__(256)
void gemm_wmma(const float* __restrict__ A, const float* __restrict__ Hn,
               const int* __restrict__ idx0, const int* __restrict__ idx1,
               const _Float16* __restrict__ Wpk, const float* __restrict__ bias,
               float* __restrict__ C, int M, int K, int Nout, int KT) {
  __shared__ __align__(32) _Float16 sA[4][32][16];  // [row-tile][wmma lane][j]

  const int tid = threadIdx.x;
  const int wave = tid >> 5;
  const int lane = tid & 31;
  const int rg = wave & 3;
  const int ng = wave >> 2;
  const int m0 = blockIdx.x * 64;
  const int n0 = blockIdx.y * 64;
  const int ntb = (n0 >> 4) + ng * 2;  // first of this wave's 2 n-tiles

  v8f acc0 = {}, acc1 = {};

  // Per-thread A-stage geometry (2 float4 per thread per stage).
  int r_[2], kk0_[2], tile_[2], slane_[2], j0_[2];
  #pragma unroll
  for (int i = 0; i < 2; ++i) {
    int f = i * 256 + tid;       // float4 id within 64x32 tile
    r_[i] = f >> 3;              // row 0..63
    kk0_[i] = (f & 7) << 2;      // k offset 0,4,...,28
    tile_[i] = r_[i] >> 4;
    slane_[i] = (r_[i] & 15) + (((kk0_[i] & 15) >= 8) ? 16 : 0);
    j0_[i] = ((kk0_[i] >> 4) << 3) + (kk0_[i] & 7);
  }

  for (int kt = 0; kt < KT; ++kt) {
    const int k0 = kt * 32;
    // ---- stage A tile (branchless clamped loads + cndmask select) ----
    #pragma unroll
    for (int i = 0; i < 2; ++i) {
      int gm = m0 + r_[i];
      int gk = k0 + kk0_[i];
      int gmc = min(gm, M - 1);
      int gkc = min(gk, K - 4);               // K is a multiple of 4
      bool valid = (gm < M) && (gk < K);
      const float* p;
      if (GATHER == 0) {
        p = A + (size_t)gmc * K + gkc;
      } else {
        int node = (gkc < D) ? idx0[gmc] : idx1[gmc];
        int kf = (gkc < D) ? gkc : gkc - D;
        p = Hn + (size_t)node * D + kf;
      }
      float4 val = *(const float4*)p;
      if (kt + 1 < KT) __builtin_prefetch(p + 32, 0, 3);  // next K-chunk
      if (!valid) { val.x = 0.f; val.y = 0.f; val.z = 0.f; val.w = 0.f; }
      v4h pk;
      pk[0] = (_Float16)val.x; pk[1] = (_Float16)val.y;
      pk[2] = (_Float16)val.z; pk[3] = (_Float16)val.w;
      *(v4h*)&sA[tile_[i]][slane_[i]][j0_[i]] = pk;  // ds_store_b64
    }
    __syncthreads();

    // ---- fragments: contiguous LDS/global 128-bit loads ----
    v16h af = *(const v16h*)&sA[rg][lane][0];  // 2x ds_load_b128
    const _Float16* wp = Wpk + (((size_t)ntb * KT + kt) * 32 + lane) * 16;
    v16h bf0 = *(const v16h*)wp;                           // 2x global_load_b128
    v16h bf1 = *(const v16h*)(wp + (size_t)KT * 32 * 16);  // next n-tile

    acc0 = __builtin_amdgcn_wmma_f32_16x16x32_f16(false, af, false, bf0,
                                                  (short)0, acc0, false, false);
    acc1 = __builtin_amdgcn_wmma_f32_16x16x32_f16(false, af, false, bf1,
                                                  (short)0, acc1, false, false);
    __syncthreads();
  }

  // ---- epilogue: bias + activation fused, ISA f32 C/D layout ----
  const int lcol = lane & 15;
  const int rbase = m0 + rg * 16 + ((lane < 16) ? 0 : 8);
  #pragma unroll
  for (int s = 0; s < 2; ++s) {
    int col = (ntb + s) * 16 + lcol;
    if (col < Nout) {
      float bv = bias[col];
      #pragma unroll
      for (int v = 0; v < 8; ++v) {
        int row = rbase + v;
        if (row < M) {
          float o = (s ? acc1[v] : acc0[v]) + bv;
          if (RELU) o = fmaxf(o, 0.f);
          C[(size_t)row * Nout + col] = o;
        }
      }
    }
  }
}

// ---------------------------------------------------------------------------
// BatchNorm column stats: per-block LDS reduction then global atomics.
// ---------------------------------------------------------------------------
__global__ void bn_stats_kernel(const float* __restrict__ X, float* __restrict__ colsum,
                                float* __restrict__ colsq, int n) {
  __shared__ float ssum[D];
  __shared__ float ssq[D];
  for (int i = threadIdx.x; i < D; i += 256) { ssum[i] = 0.f; ssq[i] = 0.f; }
  __syncthreads();
  const int ROWS = 64;
  int r0 = blockIdx.x * ROWS;
  for (int e = threadIdx.x; e < ROWS * D; e += 256) {
    int r = r0 + e / D;
    int c = e % D;
    if (r < n) {
      float v = X[(size_t)r * D + c];
      atomAddLds(&ssum[c], v);
      atomAddLds(&ssq[c], v * v);
    }
  }
  __syncthreads();
  for (int i = threadIdx.x; i < D; i += 256) {
    atomAddGlobal(&colsum[i], ssum[i]);
    atomAddGlobal(&colsq[i], ssq[i]);
  }
}

__global__ void bn_apply_kernel(const float* __restrict__ Xin, float* __restrict__ Hout,
                                const float* __restrict__ colsum, const float* __restrict__ colsq,
                                const float* __restrict__ gamma, const float* __restrict__ beta,
                                int n, int do_relu) {
  int t = blockIdx.x * blockDim.x + threadIdx.x;
  if (t >= n * D) return;
  int c = t % D;
  float inv_n = 1.0f / (float)n;
  float mean = colsum[c] * inv_n;
  float var = colsq[c] * inv_n - mean * mean;
  float v = (Xin[t] - mean) * rsqrtf(var + BN_EPS) * gamma[c] + beta[c];
  if (do_relu) v = fmaxf(v, 0.f);
  Hout[t] = v;
}

// ---------------------------------------------------------------------------
// Head final: out[q] = sigmoid(dot(U[q,:], W2) + b2)
// ---------------------------------------------------------------------------
__global__ void head_final_kernel(const float* __restrict__ U, const float* __restrict__ W2,
                                  const float* __restrict__ b2, float* __restrict__ out, int EQ) {
  int q = blockIdx.x * blockDim.x + threadIdx.x;
  if (q >= EQ) return;
  const float* u = U + (size_t)q * D;
  float s = b2[0];
  #pragma unroll 4
  for (int k = 0; k < D; ++k) s += u[k] * W2[k];
  out[q] = 1.f / (1.f + expf(-s));
}

static inline int cdiv(int a, int b) { return (a + b - 1) / b; }

extern "C" void kernel_launch(void* const* d_in, const int* in_sizes, int n_in,
                              void* d_out, int out_size, void* d_ws, size_t ws_size,
                              hipStream_t stream) {
  const int N  = in_sizes[0];
  const int E  = in_sizes[1] / 2;
  const int EQ = in_sizes[3] / 2;

  const int* edge_index = (const int*)d_in[1];
  const int* eli        = (const int*)d_in[3];
  const int* src = edge_index;
  const int* dst = edge_index + E;
  const int* q0 = eli;
  const int* q1 = eli + EQ;

  const float* node_W0 = (const float*)d_in[4];
  const float* node_b0 = (const float*)d_in[5];
  const float* node_W1 = (const float*)d_in[6];
  const float* node_b1 = (const float*)d_in[7];
  const float* node_W2 = (const float*)d_in[8];
  const float* node_b2 = (const float*)d_in[9];
  const float* edge_W0 = (const float*)d_in[10];
  const float* edge_b0 = (const float*)d_in[11];
  const float* edge_W1 = (const float*)d_in[12];
  const float* edge_b1 = (const float*)d_in[13];
  const float* edge_W2 = (const float*)d_in[14];
  const float* edge_b2 = (const float*)d_in[15];
  const float* conv_Wa = (const float*)d_in[16];  // [L, D, 2D]
  const float* conv_ba = (const float*)d_in[17];  // [L, 2D]
  const float* conv_Wb = (const float*)d_in[18];  // [L, 2D, D]
  const float* conv_bb = (const float*)d_in[19];  // [L, D]
  const float* bn_gamma = (const float*)d_in[20]; // [L, D]
  const float* bn_beta  = (const float*)d_in[21]; // [L, D]
  const float* out_W1 = (const float*)d_in[22];   // [2D, D]
  const float* out_b1 = (const float*)d_in[23];   // [D]
  const float* out_W2 = (const float*)d_in[24];   // [D, 1]
  const float* out_b2 = (const float*)d_in[25];   // [1]

  float* outp = (float*)d_out;

  // Workspace carve (offsets keep >=32B alignment).
  float* ws = (float*)d_ws;
  float* h0      = ws;              // 304
  float* e0      = ws + 304;        // 304
  float* colsum  = ws + 608;        // 304
  float* colsq   = ws + 912;        // 304
  _Float16* Wpk  = (_Float16*)(ws + 1216);    // up to 204800 f16 = 102400 floats
  float* Hbuf    = ws + 1216 + 102400;        // N*D
  float* Xbuf    = Hbuf + (size_t)N * D;      // N*D
  float* Tbuf    = Xbuf + (size_t)N * D;      // N*2D (reused as U [EQ,D])

  const int ND = N * D;

  // GEMM geometry (padded tile counts).
  const int KT_a = cdiv(D, 32);    // 10  (K=300)
  const int KT_b = cdiv(D2, 32);   // 19  (K=600)
  const int NTp_a = cdiv(D2, 64) * 4;  // 40 n-tiles (Nout=600), padded to grid
  const int NTp_b = cdiv(D, 64) * 4;   // 20 n-tiles (Nout=300)

  // 1. Encoders (constant-ones inputs collapse to a single D-vector each).
  encoder_kernel<<<1, 320, 0, stream>>>(node_W0, node_b0, node_W1, node_b1, node_W2, node_b2, h0);
  encoder_kernel<<<1, 320, 0, stream>>>(edge_W0, edge_b0, edge_W1, edge_b1, edge_W2, edge_b2, e0);

  // 2. Broadcast initial node features.
  broadcast_kernel<<<cdiv(ND, 256), 256, 0, stream>>>(h0, Hbuf, ND);

  // 3. GINE layers.
  for (int i = 0; i < NLAYERS; ++i) {
    copy_kernel<<<cdiv(ND, 256), 256, 0, stream>>>(Hbuf, Xbuf, ND);
    message_kernel<<<cdiv(E * (D / 4), 256), 256, 0, stream>>>(Hbuf, e0, src, dst, Xbuf, E);

    // T = relu(X @ Wa + ba)  [N, 600]
    {
      int tot = NTp_a * KT_a * 32;
      pack_w_kernel<<<cdiv(tot, 256), 256, 0, stream>>>(conv_Wa + (size_t)i * D * D2, Wpk,
                                                        D, D2, KT_a, NTp_a);
      dim3 g(cdiv(N, 64), cdiv(D2, 64));
      gemm_wmma<0, true><<<g, 256, 0, stream>>>(Xbuf, nullptr, nullptr, nullptr,
                                                Wpk, conv_ba + (size_t)i * D2,
                                                Tbuf, N, D, D2, KT_a);
    }
    // X = T @ Wb + bb  [N, 300]
    {
      int tot = NTp_b * KT_b * 32;
      pack_w_kernel<<<cdiv(tot, 256), 256, 0, stream>>>(conv_Wb + (size_t)i * D2 * D, Wpk,
                                                        D2, D, KT_b, NTp_b);
      dim3 g(cdiv(N, 64), cdiv(D, 64));
      gemm_wmma<0, false><<<g, 256, 0, stream>>>(Tbuf, nullptr, nullptr, nullptr,
                                                 Wpk, conv_bb + (size_t)i * D,
                                                 Xbuf, N, D2, D, KT_b);
    }
    // BatchNorm (batch stats) + relu (except last layer)
    zero_kernel<<<cdiv(608, 256), 256, 0, stream>>>(colsum, 608);
    bn_stats_kernel<<<cdiv(N, 64), 256, 0, stream>>>(Xbuf, colsum, colsq, N);
    bn_apply_kernel<<<cdiv(ND, 256), 256, 0, stream>>>(Xbuf, Hbuf, colsum, colsq,
                                                       bn_gamma + (size_t)i * D,
                                                       bn_beta + (size_t)i * D,
                                                       N, (i < NLAYERS - 1) ? 1 : 0);
  }

  // 4. Head: U = relu(concat(H[q0], H[q1]) @ W1 + b1), gather fused into A loads.
  {
    int tot = NTp_b * KT_b * 32;
    pack_w_kernel<<<cdiv(tot, 256), 256, 0, stream>>>(out_W1, Wpk, D2, D, KT_b, NTp_b);
    dim3 g(cdiv(EQ, 64), cdiv(D, 64));
    gemm_wmma<1, true><<<g, 256, 0, stream>>>(nullptr, Hbuf, q0, q1,
                                              Wpk, out_b1, Tbuf, EQ, D2, D, KT_b);
  }
  // 5. logits + sigmoid.
  head_final_kernel<<<cdiv(EQ, 256), 256, 0, stream>>>(Tbuf, out_W2, out_b2, outp, EQ);
}
